// Detect_69870527971778
// MI455X (gfx1250) — matrix operand order repaired and checked by reading, not compiled, not run on hardware
//
#include <hip/hip_runtime.h>
#include <stdint.h>

typedef __attribute__((ext_vector_type(16))) __bf16 v16bf;
typedef __attribute__((ext_vector_type(8)))  float  v8f;

__device__ __forceinline__ unsigned short f32_to_bf16_rne(float f) {
    union { float f; unsigned int u; } cv; cv.f = f;
    unsigned int u = cv.u;
    u += 0x7FFFu + ((u >> 16) & 1u);   // round-to-nearest-even
    return (unsigned short)(u >> 16);
}

__device__ __forceinline__ float sigmoidf_(float x) {
    return 1.0f / (1.0f + __expf(-x));
}

// One wave computes a 16(pos) x 64(out-ch) tile with v_wmma_f32_16x16x32_bf16.
// 8 waves / block -> 128 positions per block. N=57 real channels, padded to 64.
// C/HW/NX are template constants so A-loads use immediate offsets (one clause,
// no per-load address math) and epilogue divisions become const mul/shift.
template <int C, int HW, int NX>
__global__ __launch_bounds__(256)
void detect_head_kernel(const float* __restrict__ xin,
                        const float* __restrict__ w,
                        const float* __restrict__ bias,
                        float* __restrict__ out,
                        int scale_off, float stride,
                        float a0w, float a0h, float a1w, float a1h,
                        float a2w, float a2h)
{
    constexpr int NG = C / 16;                 // 32B bf16 granules per weight row
    __shared__ __attribute__((aligned(64))) unsigned short lds_w[64 * C];

    // ---- Stage weights fp32 -> bf16 into LDS, XOR-swizzled on 32B granules
    // so B-fragment ds reads (one granule per lane) are bank-conflict free.
    for (int i = threadIdx.x; i < 64 * C; i += blockDim.x) {
        int o = i / C, c = i - o * C;
        float v = (o < 57) ? w[o * C + c] : 0.0f;
        int g  = c >> 4;
        int sw = (((g ^ o) & (NG - 1)) << 4) | (c & 15);
        lds_w[o * C + sw] = f32_to_bf16_rne(v);
    }
    __syncthreads();

    const int lane = threadIdx.x & 31;
    const int wave = threadIdx.x >> 5;
    const int tileBase = (blockIdx.x * 8 + wave) * 16;

    // A fragment in this code's contiguous k order (16-bit A 16x32): lane holds row M = lane%16;
    // lanes 0-15 cover K 0-15, lanes 16-31 cover K 16-31.
    const int mL    = lane & 15;
    const int posL  = tileBase + mL;
    const int bL    = posL / HW;               // HW % 16 == 0 -> tile never straddles batch
    const int sL    = posL - bL * HW;

    const int kgrpA = (lane < 16) ? 0 : 16;    // A lane-group K base
    const int kgrpB = (lane < 16) ? 0 : 16;    // B lane-group K base
    const int nLane = lane & 15;               // B/C/D: N = lane%16 (+ 16*t)

    // Lane-group base pointer: all 16 per-chunk loads become immediate offsets
    // (max 15*HW*4 = 1.54 MB < +-8MB signed 24-bit ioffset).
    const float* pA = xin + ((size_t)bL * C + kgrpA) * HW + sL;

    v8f acc[4] = {v8f{}, v8f{}, v8f{}, v8f{}};

    for (int k0 = 0; k0 < C; k0 += 32) {
        const float* pc = pA + (size_t)k0 * HW;
        // ---- A fragment: 16 const-offset fp32 loads (2x 64B segments / instr)
        union { v16bf v; unsigned short u[16]; } afrag;
        #pragma unroll
        for (int e = 0; e < 16; ++e) {
            const int kk = e;                            // + kgrpA folded into pA
            afrag.u[e] = f32_to_bf16_rne(pc[(size_t)kk * HW]);
        }
        if (k0 + 32 < C)
            __builtin_prefetch(pc + (size_t)32 * HW, 0, 1);

        // ---- 4 N-tiles: B fragment = one swizzled 32B granule per lane from LDS
        const int gsrc = (k0 + kgrpB) >> 4;
        #pragma unroll
        for (int t = 0; t < 4; ++t) {
            const int orow = t * 16 + nLane;
            const int gs   = (((gsrc ^ orow) & (NG - 1)) << 4);
            const v16bf bfrag = *(const v16bf*)&lds_w[orow * C + gs];
            acc[t] = __builtin_amdgcn_wmma_f32_16x16x32_bf16(
                false, afrag.v, false, bfrag, (short)0, acc[t], false, false);
        }
    }

    // ---- Element-wise decode straight from accumulators.
    // C/D layout: VGPR r -> M = r (+8 for lanes 16-31); N = t*16 + lane%16.
    const float anchw[3] = {a0w, a1w, a2w};
    const float anchh[3] = {a0h, a1h, a2h};
    #pragma unroll
    for (int t = 0; t < 4; ++t) {
        const int N = t * 16 + nLane;
        if (N >= 57) continue;                 // padded channels
        const int   aIdx = N / 19;             // anchor
        const int   oCh  = N - aIdx * 19;      // channel within anchor
        const float bv   = bias[N];
        const float aw   = anchw[aIdx], ah = anchh[aIdx];
        #pragma unroll
        for (int r = 0; r < 8; ++r) {
            const int M   = r + ((lane < 16) ? 0 : 8);
            const int pos = tileBase + M;
            const int bb  = pos / HW;          // const divisor -> mul/shift
            const int s   = pos - bb * HW;
            const int yi  = s / NX;
            const int xi  = s - yi * NX;
            const float p = acc[t][r] + bv;
            float res;
            if (oCh < 2) {                                   // xy
                const float g = (oCh == 0) ? (float)xi : (float)yi;
                res = (sigmoidf_(p) * 2.0f - 0.5f + g) * stride;
            } else if (oCh < 4) {                            // wh
                const float ss = sigmoidf_(p) * 2.0f;
                res = ss * ss * ((oCh == 2) ? aw : ah);
            } else if (oCh == 4) {                           // conf
                res = sigmoidf_(p);
            } else if (oCh < 17) {                           // landmarks (raw logit)
                const int j = oCh - 5;
                res = p * ((j & 1) ? ah : aw)
                    + ((j & 1) ? (float)yi : (float)xi) * stride;
            } else {                                         // cls
                res = sigmoidf_(p);
            }
            out[((size_t)bb * 100800u + (size_t)scale_off
                 + (size_t)aIdx * HW + s) * 19 + oCh] = res;
        }
    }
}

extern "C" void kernel_launch(void* const* d_in, const int* in_sizes, int n_in,
                              void* d_out, int out_size, void* d_ws, size_t ws_size,
                              hipStream_t stream) {
    const float* x0 = (const float*)d_in[0];
    const float* x1 = (const float*)d_in[1];
    const float* x2 = (const float*)d_in[2];
    const float* w0 = (const float*)d_in[3];
    const float* w1 = (const float*)d_in[4];
    const float* w2 = (const float*)d_in[5];
    const float* b0 = (const float*)d_in[6];
    const float* b1 = (const float*)d_in[7];
    const float* b2 = (const float*)d_in[8];
    float* out = (float*)d_out;
    dim3 blk(256);
    // blocks = 16*hw/128 per scale; row offsets: 0, 3*25600, 3*25600+3*6400
    detect_head_kernel<128, 25600, 160><<<3200, blk, 0, stream>>>(
        x0, w0, b0, out,     0,  8.f, 10.f, 13.f, 16.f, 30.f, 33.f, 23.f);
    detect_head_kernel<256,  6400,  80><<< 800, blk, 0, stream>>>(
        x1, w1, b1, out, 76800, 16.f, 30.f, 61.f, 62.f, 45.f, 59.f, 119.f);
    detect_head_kernel<512,  1600,  40><<< 200, blk, 0, stream>>>(
        x2, w2, b2, out, 96000, 32.f, 116.f, 90.f, 156.f, 198.f, 373.f, 326.f);
}
